// HypE_16552803959289
// MI455X (gfx1250) — matrix operand order repaired and checked by reading, not compiled, not run on hardware
//
#include <hip/hip_runtime.h>
#include <hip/hip_bf16.h>

// ---------------- problem constants ----------------
#define BATCH   1024
#define D1      200
#define OUTC    32
#define RDIM    288        // 1*32*1*9
#define SPAT    120        // 10*12
#define FCK     3840       // 32*10*12
#define NENT    100000
#define KPAD    224        // D1=200 padded to 7*32 for bf16 WMMA K-steps
#define NPADFC  208        // 200 padded to 13*16 N-tiles for the FC GEMM
#define EPS     1e-5f

// ---------------- WMMA types ----------------
typedef __attribute__((ext_vector_type(16))) __bf16 v16bf;
typedef __attribute__((ext_vector_type(8)))  float  v8f;

union Frag { v16bf v; uint4 q[2]; };

__device__ __forceinline__ unsigned short f2bf(float f) {
    unsigned int x = __float_as_uint(f);
    return (unsigned short)((x + 0x7FFFu + ((x >> 16) & 1u)) >> 16);  // RNE
}

// ---------------- K1: gather e1 embeddings + BN0 partial stats ----------------
__global__ void k_gather_stats0(const int* __restrict__ e1_idx,
                                const float* __restrict__ E_w,
                                float* __restrict__ g, float* __restrict__ part) {
    __shared__ float s1[256], s2[256];
    int tid = threadIdx.x;
    float s = 0.f, ss = 0.f;
    for (int i = blockIdx.x * 256 + tid; i < BATCH * D1; i += gridDim.x * 256) {
        int b = i / D1, k = i % D1;
        float v = E_w[(size_t)e1_idx[b] * D1 + k];
        g[i] = v; s += v; ss += v * v;
    }
    s1[tid] = s; s2[tid] = ss; __syncthreads();
    for (int o = 128; o > 0; o >>= 1) {
        if (tid < o) { s1[tid] += s1[tid + o]; s2[tid] += s2[tid + o]; }
        __syncthreads();
    }
    if (tid == 0) { part[2 * blockIdx.x] = s1[0]; part[2 * blockIdx.x + 1] = s2[0]; }
}

// ---------------- finalize BN0 -> affine (a0, d0) ----------------
__global__ void k_fin0(const float* __restrict__ part,
                       const float* __restrict__ g0, const float* __restrict__ b0,
                       float* __restrict__ ab0) {
    if (threadIdx.x == 0) {
        float s = 0.f, ss = 0.f;
        for (int i = 0; i < 64; i++) { s += part[2 * i]; ss += part[2 * i + 1]; }
        float n = (float)(BATCH * D1);
        float mean = s / n, var = ss / n - mean * mean;
        float a = g0[0] * rsqrtf(var + EPS);
        ab0[0] = a; ab0[1] = b0[0] - mean * a;
    }
}

// ---------------- per-sample dynamic 1x9 VALID conv ----------------
__global__ void k_conv(const float* __restrict__ g, const int* __restrict__ r_idx,
                       const float* __restrict__ R_w, const float* __restrict__ ab0,
                       float* __restrict__ y) {
    __shared__ float xs[D1];    // normalized 10x20 image
    __shared__ float ks[RDIM];  // 32 filters of 9 taps
    int b = blockIdx.x, t = threadIdx.x;
    float a0 = ab0[0], d0 = ab0[1];
    for (int i = t; i < D1;   i += 128) xs[i] = a0 * g[b * D1 + i] + d0;
    const float* rp = R_w + (size_t)r_idx[b] * RDIM;
    for (int i = t; i < RDIM; i += 128) ks[i] = rp[i];
    __syncthreads();
    for (int o = t; o < FCK; o += 128) {
        int c = o / SPAT, s = o % SPAT, oh = s / 12, ow = s % 12;
        const float* xr = xs + oh * 20 + ow;
        const float* kr = ks + c * 9;
        float acc = 0.f;
#pragma unroll
        for (int f = 0; f < 9; f++) acc += xr[f] * kr[f];
        y[(size_t)b * FCK + o] = acc;
    }
}

// ---------------- BN1 stats: one block per channel, finalize inline ----------------
__global__ void k_stats1(const float* __restrict__ y,
                         const float* __restrict__ g1, const float* __restrict__ b1,
                         float* __restrict__ a1, float* __restrict__ d1) {
    __shared__ float s1[256], s2[256];
    int c = blockIdx.x, tid = threadIdx.x;
    float s = 0.f, ss = 0.f;
    for (int i = tid; i < BATCH * SPAT; i += 256) {
        int b = i / SPAT, sp = i % SPAT;
        float v = y[(size_t)b * FCK + c * SPAT + sp];
        s += v; ss += v * v;
    }
    s1[tid] = s; s2[tid] = ss; __syncthreads();
    for (int o = 128; o > 0; o >>= 1) {
        if (tid < o) { s1[tid] += s1[tid + o]; s2[tid] += s2[tid + o]; }
        __syncthreads();
    }
    if (tid == 0) {
        float n = (float)(BATCH * SPAT);
        float mean = s1[0] / n, var = s2[0] / n - mean * mean;
        float a = g1[c] * rsqrtf(var + EPS);
        a1[c] = a; d1[c] = b1[c] - mean * a;
    }
}

// ---------------- apply BN1 + cvt bf16 ----------------
__global__ void k_ybf(const float* __restrict__ y, const float* __restrict__ a1,
                      const float* __restrict__ d1, unsigned short* __restrict__ ybf) {
    for (int i = blockIdx.x * 256 + threadIdx.x; i < BATCH * FCK; i += gridDim.x * 256) {
        int c = (i % FCK) / SPAT;
        ybf[i] = f2bf(a1[c] * y[i] + d1[c]);
    }
}

// ---------------- fc_w -> bf16, N padded to 208 ----------------
__global__ void k_fcwbf(const float* __restrict__ fcw, unsigned short* __restrict__ w) {
    for (int i = blockIdx.x * 256 + threadIdx.x; i < NPADFC * FCK; i += gridDim.x * 256) {
        int r = i / FCK, k = i % FCK;
        w[i] = (r < D1) ? f2bf(fcw[r * FCK + k]) : (unsigned short)0;
    }
}

// ---------------- FC GEMM: [1024,3840] x [3840,200] via bf16 WMMA ----------------
// One wave per 16x16 output tile; NT layout, both operands K-contiguous.
__global__ __launch_bounds__(32) void k_fc_gemm(const unsigned short* __restrict__ ybf,
                                                const unsigned short* __restrict__ wbf,
                                                const float* __restrict__ fc_b,
                                                float* __restrict__ fco) {
    int lane = threadIdx.x & 31;
    int lm = lane & 15, hi = (lane >> 4) & 1;
    int m0 = blockIdx.y * 16, n0 = blockIdx.x * 16;
    const unsigned short* ap = ybf + (size_t)(m0 + lm) * FCK + hi * 8;   // A: interleaved K split
    const unsigned short* bp = wbf + (size_t)(n0 + lm) * FCK + hi * 8;   // B: interleaved K split
    v8f acc = {};
    for (int k = 0; k < FCK; k += 32) {
        Frag a, b;
        a.q[0] = *(const uint4*)(ap + k);
        a.q[1] = *(const uint4*)(ap + k + 16);
        b.q[0] = *(const uint4*)(bp + k);
        b.q[1] = *(const uint4*)(bp + k + 16);
        acc = __builtin_amdgcn_wmma_f32_16x16x32_bf16(false, a.v, false, b.v,
                                                      (short)0, acc, false, false);
    }
    int n = n0 + lm;
    if (n < D1) {
        float bias = fc_b[n];
#pragma unroll
        for (int r = 0; r < 8; r++) {
            int m = m0 + r + hi * 8;
            fco[(size_t)m * D1 + n] = acc[r] + bias;
        }
    }
}

// ---------------- BN2 stats: one block per feature ----------------
__global__ void k_stats2(const float* __restrict__ fco,
                         const float* __restrict__ g2, const float* __restrict__ b2,
                         float* __restrict__ a2, float* __restrict__ d2) {
    __shared__ float s1[256], s2[256];
    int j = blockIdx.x, tid = threadIdx.x;
    float s = 0.f, ss = 0.f;
    for (int b = tid; b < BATCH; b += 256) {
        float v = fco[(size_t)b * D1 + j];
        s += v; ss += v * v;
    }
    s1[tid] = s; s2[tid] = ss; __syncthreads();
    for (int o = 128; o > 0; o >>= 1) {
        if (tid < o) { s1[tid] += s1[tid + o]; s2[tid] += s2[tid + o]; }
        __syncthreads();
    }
    if (tid == 0) {
        float n = (float)BATCH;
        float mean = s1[0] / n, var = s2[0] / n - mean * mean;
        float a = g2[j] * rsqrtf(var + EPS);
        a2[j] = a; d2[j] = b2[j] - mean * a;
    }
}

// ---------------- BN2 + ReLU + cvt bf16, K padded to 224 ----------------
__global__ void k_zbf(const float* __restrict__ fco, const float* __restrict__ a2,
                      const float* __restrict__ d2, unsigned short* __restrict__ zbf) {
    for (int i = blockIdx.x * 256 + threadIdx.x; i < BATCH * KPAD; i += gridDim.x * 256) {
        int b = i / KPAD, k = i % KPAD;
        unsigned short v = 0;
        if (k < D1) {
            float t = a2[k] * fco[(size_t)b * D1 + k] + d2[k];
            v = f2bf(t > 0.f ? t : 0.f);
        }
        zbf[i] = v;
    }
}

// ---------------- E_w -> bf16, K padded to 224 ----------------
__global__ void k_ebf(const float* __restrict__ E, unsigned short* __restrict__ ebf) {
    for (size_t i = (size_t)blockIdx.x * 256 + threadIdx.x; i < (size_t)NENT * KPAD;
         i += (size_t)gridDim.x * 256) {
        size_t n = i / KPAD; int k = (int)(i % KPAD);
        ebf[i] = (k < D1) ? f2bf(E[n * D1 + k]) : (unsigned short)0;
    }
}

// ---------------- FINAL GEMM: [1024,224bf16] x E^T -> sigmoid([1024,100000]) ----------------
// 256 threads = 8 waves arranged 2(M) x 4(N); block tile 128x128; wave tile 64M x 32N.
// A tile (Z) staged in LDS (56 KB) -> ds_load_b128 fragments; B fragments from L2-resident
// bf16 E (only 2 waves share each N-strip -> half the redundant L2 traffic of a 4x2 grid).
__global__ __launch_bounds__(256) void k_score_gemm(const unsigned short* __restrict__ zbf,
                                                    const unsigned short* __restrict__ ebf,
                                                    const float* __restrict__ b_bias,
                                                    float* __restrict__ out) {
    __shared__ unsigned short As[128 * KPAD];  // 57344 bytes
    int tid = threadIdx.x;
    int n0 = blockIdx.x * 128, m0 = blockIdx.y * 128;

    // Stage the 128x224 bf16 A tile (rows are contiguous -> pure linear copy).
    {
        const uint4* src = (const uint4*)(zbf + (size_t)m0 * KPAD);
        uint4* dst = (uint4*)As;
        for (int i = tid; i < (128 * KPAD) / 8; i += 256) dst[i] = src[i];
    }
    __syncthreads();

    int w = tid >> 5, lane = tid & 31;
    int wm = w & 1, wn = w >> 1;           // wave grid 2(M) x 4(N)
    int lm = lane & 15, hi = (lane >> 4) & 1;
    int nb = n0 + wn * 32;
    int mbase = wm * 64;

    v8f acc[4][2];
#pragma unroll
    for (int i = 0; i < 4; i++)
#pragma unroll
        for (int j = 0; j < 2; j++) acc[i][j] = {};

    for (int k0 = 0; k0 < KPAD; k0 += 32) {
        Frag bfr[2];
#pragma unroll
        for (int j = 0; j < 2; j++) {
            int n = nb + j * 16 + lm;
            n = n < NENT ? n : NENT - 1;                  // clamp; result masked at store
            const unsigned short* bp = ebf + (size_t)n * KPAD + k0 + hi * 8;
            bfr[j].q[0] = *(const uint4*)bp;              // B: interleaved K split
            bfr[j].q[1] = *(const uint4*)(bp + 16);
        }
#pragma unroll
        for (int i = 0; i < 4; i++) {
            Frag afr;
            const unsigned short* apl =
                As + (mbase + i * 16 + lm) * KPAD + k0 + hi * 8;  // A: interleaved K split
            afr.q[0] = *(const uint4*)apl;
            afr.q[1] = *(const uint4*)(apl + 16);
#pragma unroll
            for (int j = 0; j < 2; j++)
                acc[i][j] = __builtin_amdgcn_wmma_f32_16x16x32_bf16(
                    false, afr.v, false, bfr[j].v, (short)0, acc[i][j], false, false);
        }
    }

    // Epilogue: + b_bias, sigmoid, store. C/D layout: lane<16 -> M=r, lane>=16 -> M=r+8; N=lane%16.
#pragma unroll
    for (int j = 0; j < 2; j++) {
        int n = nb + j * 16 + lm;
        if (n < NENT) {
            float bias = b_bias[n];
#pragma unroll
            for (int i = 0; i < 4; i++) {
#pragma unroll
                for (int r = 0; r < 8; r++) {
                    int m = m0 + mbase + i * 16 + r + hi * 8;
                    float v = acc[i][j][r] + bias;
                    out[(size_t)m * NENT + n] = 1.0f / (1.0f + __expf(-v));
                }
            }
        }
    }
}

// ---------------- workspace layout (bytes) ----------------
#define OFF_G     ((size_t)0)                    // 1024*200*4      =   819200
#define OFF_Y     ((size_t)819200)               // 1024*3840*4     = 15728640
#define OFF_FC    ((size_t)16547840)             // 1024*200*4      =   819200
#define OFF_ST    ((size_t)17367040)             // stats block       (2560)
#define OFF_YBF   ((size_t)17369600)             // 1024*3840*2     =  7864320
#define OFF_WBF   ((size_t)25233920)             // 208*3840*2      =  1597440
#define OFF_ZBF   ((size_t)26831360)             // 1024*224*2      =   458752
#define OFF_EBF   ((size_t)27290112)             // 100000*224*2    = 44800000
// total ~72.1 MB

extern "C" void kernel_launch(void* const* d_in, const int* in_sizes, int n_in,
                              void* d_out, int out_size, void* d_ws, size_t ws_size,
                              hipStream_t stream) {
    const int*   e1_idx = (const int*)d_in[0];
    const int*   r_idx  = (const int*)d_in[1];
    const float* E_w    = (const float*)d_in[2];
    const float* R_w    = (const float*)d_in[3];
    const float* fc_w   = (const float*)d_in[4];
    const float* fc_b   = (const float*)d_in[5];
    const float* bn0_g  = (const float*)d_in[6];
    const float* bn0_b  = (const float*)d_in[7];
    const float* bn1_g  = (const float*)d_in[8];
    const float* bn1_b  = (const float*)d_in[9];
    const float* bn2_g  = (const float*)d_in[10];
    const float* bn2_b  = (const float*)d_in[11];
    const float* b_bias = (const float*)d_in[12];
    float* out = (float*)d_out;

    char* ws = (char*)d_ws;
    float* g    = (float*)(ws + OFF_G);
    float* y    = (float*)(ws + OFF_Y);
    float* fco  = (float*)(ws + OFF_FC);
    float* st   = (float*)(ws + OFF_ST);
    unsigned short* ybf = (unsigned short*)(ws + OFF_YBF);
    unsigned short* wbf = (unsigned short*)(ws + OFF_WBF);
    unsigned short* zbf = (unsigned short*)(ws + OFF_ZBF);
    unsigned short* ebf = (unsigned short*)(ws + OFF_EBF);

    float* part = st;        // [128] per-block partial sums for BN0
    float* ab0  = st + 128;  // [2]   BN0 affine
    float* a1   = st + 132;  // [32]
    float* d1   = st + 164;  // [32]
    float* a2   = st + 200;  // [200]
    float* d2   = st + 400;  // [200]

    // Independent prep (E/fc_w bf16 conversion) launched early to overlap in-order.
    k_ebf<<<4096, 256, 0, stream>>>(E_w, ebf);
    k_fcwbf<<<1024, 256, 0, stream>>>(fc_w, wbf);

    k_gather_stats0<<<64, 256, 0, stream>>>(e1_idx, E_w, g, part);
    k_fin0<<<1, 64, 0, stream>>>(part, bn0_g, bn0_b, ab0);
    k_conv<<<BATCH, 128, 0, stream>>>(g, r_idx, R_w, ab0, y);
    k_stats1<<<OUTC, 256, 0, stream>>>(y, bn1_g, bn1_b, a1, d1);
    k_ybf<<<2048, 256, 0, stream>>>(y, a1, d1, ybf);
    k_fc_gemm<<<dim3(NPADFC / 16, BATCH / 16), 32, 0, stream>>>(ybf, wbf, fc_b, fco);
    k_stats2<<<D1, 256, 0, stream>>>(fco, bn2_g, bn2_b, a2, d2);
    k_zbf<<<896, 256, 0, stream>>>(fco, a2, d2, zbf);
    k_score_gemm<<<dim3((NENT + 127) / 128, BATCH / 128), 256, 0, stream>>>(zbf, ebf, b_bias, out);
}
